// MobileNet_1357209666009
// MI455X (gfx1250) — matrix operand order repaired and checked by reading, not compiled, not run on hardware
//
#include <hip/hip_runtime.h>
#include <hip/hip_bf16.h>
#include <stdint.h>

// ---------------------------------------------------------------------------
// MobileNet-v1 forward for gfx1250 (MI455X).
// Pointwise convs + FC -> bf16 WMMA GEMM (v_wmma_f32_16x16x32_bf16, wave32),
// A tile staged to LDS via gfx1250 async-to-LDS (ASYNCcnt) when available,
// B pre-packed to per-lane fragment order (2x global_load_b128 per fragment).
// Everything else (conv1, depthwise, BN stats/apply) -> streaming VALU kernels.
// Activations are channels-last [N*H*W, C] so 1x1 conv == row-major GEMM.
// ---------------------------------------------------------------------------

typedef __attribute__((ext_vector_type(16))) __bf16 bf16x16;
typedef __attribute__((ext_vector_type(8)))  float  f32x8;
typedef int v4i __attribute__((vector_size(16)));   // builtin's pointee type

union FragB16 { bf16x16 v; uint4 q[2]; };

#if defined(__gfx1250__) && __has_builtin(__builtin_amdgcn_global_load_async_to_lds_b128)
#define HAVE_ASYNC_LDS 1
#else
#define HAVE_ASYNC_LDS 0
#endif

__device__ static inline void wait_asynccnt0() {
#if defined(__gfx1250__)
#if __has_builtin(__builtin_amdgcn_s_wait_asynccnt)
  __builtin_amdgcn_s_wait_asynccnt(0);
#else
  asm volatile("s_wait_asynccnt 0x0" ::: "memory");
#endif
#endif
}

// ---------------------------------------------------------------------------
// conv1: 3x3, 3->32, stride1 pad1, NCHW input -> channels-last [N*H*W, 32] f32
// ---------------------------------------------------------------------------
__global__ void k_conv1(const float* __restrict__ x, const float* __restrict__ w,
                        const float* __restrict__ b, float* __restrict__ out,
                        int N, int H, int W)
{
  const size_t idx = (size_t)blockIdx.x * blockDim.x + threadIdx.x;
  const size_t total = (size_t)N * H * W * 32;
  if (idx >= total) return;
  const int co = (int)(idx & 31);
  size_t p = idx >> 5;
  const int wo = (int)(p % W); p /= W;
  const int ho = (int)(p % H);
  const int n  = (int)(p / H);
  float acc = b[co];
  #pragma unroll
  for (int ci = 0; ci < 3; ++ci)
    #pragma unroll
    for (int dh = 0; dh < 3; ++dh) {
      const int h = ho + dh - 1;
      if (h < 0 || h >= H) continue;
      #pragma unroll
      for (int dw = 0; dw < 3; ++dw) {
        const int ww = wo + dw - 1;
        if (ww < 0 || ww >= W) continue;
        acc += w[((co * 3 + ci) * 3 + dh) * 3 + dw] *
               x[((size_t)(n * 3 + ci) * H + h) * W + ww];
      }
    }
  out[idx] = acc;
}

// ---------------------------------------------------------------------------
// Depthwise 3x3, single shared kernel (broadcast over channels), channels-last.
// C is a power of two -> mask/shift addressing.
// ---------------------------------------------------------------------------
__global__ void k_dwconv(const float* __restrict__ in, float* __restrict__ out,
                         const float* __restrict__ k9, const float* __restrict__ b1,
                         int N, int H, int W, int Ho, int Wo, int cShift, int stride)
{
  const size_t idx = (size_t)blockIdx.x * blockDim.x + threadIdx.x;
  const size_t total = ((size_t)N * Ho * Wo) << cShift;
  if (idx >= total) return;
  const int c = (int)(idx & ((1u << cShift) - 1));
  size_t p = idx >> cShift;
  const int wo = (int)(p % Wo); p /= Wo;
  const int ho = (int)(p % Ho);
  const int n  = (int)(p / Ho);
  float acc = b1[0];
  #pragma unroll
  for (int dh = 0; dh < 3; ++dh) {
    const int h = ho * stride + dh - 1;
    if (h < 0 || h >= H) continue;
    #pragma unroll
    for (int dw = 0; dw < 3; ++dw) {
      const int w = wo * stride + dw - 1;
      if (w < 0 || w >= W) continue;
      acc += k9[dh * 3 + dw] * in[(((size_t)(n * H + h) * W + w) << cShift) + c];
    }
  }
  out[idx] = acc;
}

// ---------------------------------------------------------------------------
// BN: per-channel column sums over [M, C] — fixed 64-partition deterministic tree
// ---------------------------------------------------------------------------
__global__ void k_col_stats(const float* __restrict__ X, float* __restrict__ psum,
                            float* __restrict__ psq, int M, int C)
{
  const int c = blockIdx.x * blockDim.x + threadIdx.x;
  if (c >= C) return;
  float s = 0.f, q = 0.f;
  for (int r = blockIdx.y; r < M; r += gridDim.y) {
    const float v = X[(size_t)r * C + c];
    s += v; q += v * v;
  }
  psum[blockIdx.y * C + c] = s;
  psq [blockIdx.y * C + c] = q;
}

__global__ void k_bn_finalize(const float* __restrict__ psum, const float* __restrict__ psq,
                              const float* __restrict__ g, const float* __restrict__ b,
                              float* __restrict__ scale, float* __restrict__ shift,
                              int C, float invM)
{
  const int c = blockIdx.x * blockDim.x + threadIdx.x;
  if (c >= C) return;
  float s = 0.f, q = 0.f;
  for (int p = 0; p < 64; ++p) { s += psum[p * C + c]; q += psq[p * C + c]; }
  const float mean = s * invM;
  const float var  = q * invM - mean * mean;
  const float sc   = g[c] * rsqrtf(var + 1e-5f);
  scale[c] = sc;
  shift[c] = b[c] - mean * sc;
}

__global__ void k_bnrelu_f32(float* __restrict__ X, const float* __restrict__ scale,
                             const float* __restrict__ shift, size_t total, int cMask)
{
  const size_t i = (size_t)blockIdx.x * blockDim.x + threadIdx.x;
  if (i >= total) return;
  const int c = (int)(i & (size_t)cMask);
  X[i] = fmaxf(0.f, X[i] * scale[c] + shift[c]);
}

__global__ void k_bnrelu_bf16(const float* __restrict__ X, __bf16* __restrict__ Y,
                              const float* __restrict__ scale, const float* __restrict__ shift,
                              size_t total, int cMask)
{
  const size_t i = (size_t)blockIdx.x * blockDim.x + threadIdx.x;
  if (i >= total) return;
  const int c = (int)(i & (size_t)cMask);
  Y[i] = (__bf16)fmaxf(0.f, X[i] * scale[c] + shift[c]);
}

// ---------------------------------------------------------------------------
// Pack weights W[n*K+k] (f32) into the per-lane B-fragment order for
// v_wmma_f32_16x16x32_bf16:  Bp[kt][nt][lane][e]  with
//   k = kt*32 + 16*(lane>>4) + e   (e = 2j+p <-> VGPR j half p)
//   n = nt*16 + (lane&15)
// Each wave then loads its 32-byte fragment with two global_load_b128.
// ---------------------------------------------------------------------------
__global__ void k_pack_b(const float* __restrict__ W, __bf16* __restrict__ Bp,
                         int K, int Npad, int Nreal)
{
  const size_t idx = (size_t)blockIdx.x * blockDim.x + threadIdx.x;
  const size_t total = (size_t)(K >> 5) * (Npad >> 4) * 512;
  if (idx >= total) return;
  const int e    = (int)(idx & 15);
  const int lane = (int)((idx >> 4) & 31);
  const size_t t = idx >> 9;
  const int nt = (int)(t % (size_t)(Npad >> 4));
  const int kt = (int)(t / (size_t)(Npad >> 4));
  const int k = (kt << 5) + ((lane >> 4) << 4) + e;
  const int n = (nt << 4) + (lane & 15);
  const float v = (n < Nreal) ? W[(size_t)n * K + k] : 0.0f;
  Bp[idx] = (__bf16)v;
}

// ---------------------------------------------------------------------------
// WMMA GEMM: C[M, Nreal] = A[M, K](bf16) * Bpacked(bf16) + bias, f32 accum.
// Block = 128 threads = 4 waves; block tile = 32(M) x 64(N); K-step = 32.
// Each wave owns two 16x16 M-tiles over one 16-wide N-tile: 2 WMMA per k-step,
// sharing one B fragment. A tile (32x32 bf16 = 2KB) staged in LDS, loaded with
// one async b128 (or coop b128) per thread.
// Requires M % 32 == 0, K % 32 == 0, Npad % 64 == 0 (true for every GEMM here).
// ---------------------------------------------------------------------------
__global__ void __launch_bounds__(128)
k_gemm_bf16(const __bf16* __restrict__ A, const __bf16* __restrict__ Bp,
            const float* __restrict__ bias, float* __restrict__ C,
            int M, int K, int Npad, int Nreal)
{
  __shared__ __align__(16) __bf16 Alds[32][32];
  const int tid  = threadIdx.x;
  const int lane = tid & 31;
  const int wave = tid >> 5;
  const int m0 = blockIdx.x << 5;            // 32 rows per block
  const int tn = (blockIdx.y << 2) + wave;   // 16-wide N tile index
  const int ntiles = Npad >> 4;
  const int g  = lane >> 4;                  // lane group (K-half select)
  const int ml = lane & 15;                  // A row / B col within tile

  // cooperative A-tile mapping: 128 thr x 16B = 2KB
  const int lr = tid >> 2;
  const int lc = (tid & 3) << 3;             // half offset 0,8,16,24

  f32x8 acc0 = {0.f, 0.f, 0.f, 0.f, 0.f, 0.f, 0.f, 0.f};
  f32x8 acc1 = acc0;

  for (int k0 = 0; k0 < K; k0 += 32) {
    const __bf16* gsrc = A + (size_t)(m0 + lr) * K + k0 + lc;
    __syncthreads();                          // all waves done reading Alds
#if HAVE_ASYNC_LDS
    __builtin_amdgcn_global_load_async_to_lds_b128(
        (v4i*)gsrc, (v4i*)&Alds[lr][lc], 0, 0);
    wait_asynccnt0();
#else
    *(uint4*)&Alds[lr][lc] = *(const uint4*)gsrc;
#endif
    if (k0 + 32 < K)                          // pull next A tile toward caches
      __builtin_prefetch(gsrc + 32, 0, 3);
    __syncthreads();                          // tile visible to all waves

    // B fragment: 32 contiguous bytes per lane from the packed tile
    FragB16 b, a0, a1;
    const __bf16* bp = Bp + ((size_t)(k0 >> 5) * ntiles + tn) * 512 + lane * 16;
    b.q[0] = *(const uint4*)(bp);
    b.q[1] = *(const uint4*)(bp + 8);

    // A fragments (contiguous k order, matching packed B): lanes 0-15 k=[0..15], 16-31 k=[16..31]
    a0.q[0] = *(const uint4*)&Alds[ml][16 * g];
    a0.q[1] = *(const uint4*)&Alds[ml][16 * g + 8];
    a1.q[0] = *(const uint4*)&Alds[16 + ml][16 * g];
    a1.q[1] = *(const uint4*)&Alds[16 + ml][16 * g + 8];

    acc0 = __builtin_amdgcn_wmma_f32_16x16x32_bf16(
        false, a0.v, false, b.v, (short)0, acc0, false, false);
    acc1 = __builtin_amdgcn_wmma_f32_16x16x32_bf16(
        false, a1.v, false, b.v, (short)0, acc1, false, false);
  }

  // C/D layout: VGPR e -> row e + 8*g, col = lane&15
  const int col = (tn << 4) + ml;
  if (col < Nreal) {
    const float bv = bias[col];
    #pragma unroll
    for (int e = 0; e < 8; ++e) {
      const int row = m0 + (g << 3) + e;
      C[(size_t)row * Nreal + col]        = acc0[e] + bv;
      C[(size_t)(row + 16) * Nreal + col] = acc1[e] + bv;
    }
  }
}

// ---------------------------------------------------------------------------
// Global average pool [N, 49, 1024] f32 -> [N, 1024] bf16 (GEMM A for the FC)
// ---------------------------------------------------------------------------
__global__ void k_avgpool(const float* __restrict__ X, __bf16* __restrict__ Y,
                          int N, int HW)
{
  const int idx = blockIdx.x * blockDim.x + threadIdx.x;
  if (idx >= (N << 10)) return;
  const int c = idx & 1023;
  const int n = idx >> 10;
  float s = 0.f;
  for (int p = 0; p < HW; ++p) s += X[(((size_t)n * HW + p) << 10) + c];
  Y[idx] = (__bf16)(s / (float)HW);
}

// ---------------------------------------------------------------------------
// Host orchestration
// ---------------------------------------------------------------------------
static const int CH[14] = {32,64,128,128,256,256,512,512,512,512,512,512,1024,1024};
static const int ST[13] = {1,2,1,2,1,2,1,1,1,1,1,2,2};

extern "C" void kernel_launch(void* const* d_in, const int* in_sizes, int n_in,
                              void* d_out, int out_size, void* d_ws, size_t ws_size,
                              hipStream_t stream)
{
  (void)out_size; (void)ws_size;
  const int X_ELEMS = 32 * 3 * 224 * 224;

  // ---- map inputs: detect insertion-order vs jax-sorted-leaves flattening ----
  const float *x, *conv1_w, *conv1_b, *bn0_g, *bn0_b, *fc_w, *fc_b;
  const float *dw_w[13], *dw_b[13], *bn1_g[13], *bn1_b[13];
  const float *pw_w[13], *pw_b[13], *bn2_g[13], *bn2_b[13];
  auto F = [&](int i) { return (const float*)d_in[i]; };

  if (n_in > 0 && in_sizes[0] == X_ELEMS) {
    // insertion order: x, conv1_w, conv1_b, bn0_g, bn0_b, fc_w, fc_b, blocks...
    x = F(0); conv1_w = F(1); conv1_b = F(2); bn0_g = F(3); bn0_b = F(4);
    fc_w = F(5); fc_b = F(6);
    for (int i = 0; i < 13; ++i) {
      const int b = 7 + 8 * i;
      dw_w[i] = F(b + 0); dw_b[i] = F(b + 1);
      bn1_g[i] = F(b + 2); bn1_b[i] = F(b + 3);
      pw_w[i] = F(b + 4); pw_b[i] = F(b + 5);
      bn2_g[i] = F(b + 6); bn2_b[i] = F(b + 7);
    }
  } else {
    // jax tree_leaves (sorted keys): blocks[ bn1_b,bn1_g,bn2_b,bn2_g,dw_b,dw_w,pw_b,pw_w ],
    // then bn0_b, bn0_g, conv1_b, conv1_w, fc_b, fc_w, x
    for (int i = 0; i < 13; ++i) {
      const int b = 8 * i;
      bn1_b[i] = F(b + 0); bn1_g[i] = F(b + 1);
      bn2_b[i] = F(b + 2); bn2_g[i] = F(b + 3);
      dw_b[i] = F(b + 4); dw_w[i] = F(b + 5);
      pw_b[i] = F(b + 6); pw_w[i] = F(b + 7);
    }
    bn0_b = F(104); bn0_g = F(105); conv1_b = F(106); conv1_w = F(107);
    fc_b = F(108); fc_w = F(109); x = F(110);
  }

  // ---- workspace arena (two ping-pong activation slots + small buffers) ----
  char* base = (char*)d_ws;
  size_t off = 0;
  auto alloc = [&](size_t bytes) -> char* {
    char* p = base + off;
    off += (bytes + 255) & ~(size_t)255;
    return p;
  };
  const size_t SLOT = (size_t)1605632 * 64 * 4;   // largest buffer: block-0 GEMM out
  float*  slot0  = (float*)alloc(SLOT);
  float*  slot1  = (float*)alloc(SLOT);
  float*  psum   = (float*)alloc((size_t)64 * 1024 * 4);
  float*  psq    = (float*)alloc((size_t)64 * 1024 * 4);
  float*  scale  = (float*)alloc(1024 * 4);
  float*  shift  = (float*)alloc(1024 * 4);
  __bf16* packB  = (__bf16*)alloc((size_t)32 * 64 * 512 * 2);   // K/32 * N/16 * 512 halves
  __bf16* pooled = (__bf16*)alloc((size_t)32 * 1024 * 2);

  auto cdiv = [](size_t a, size_t b) { return (unsigned)((a + b - 1) / b); };
  auto log2i = [](int v) { int s = 0; while ((1 << s) < v) ++s; return s; };

  auto stats_bn = [&](const float* X, int Mrows, int Cc,
                      const float* gam, const float* bet) {
    dim3 gs(cdiv(Cc, 256), 64);
    k_col_stats<<<gs, 256, 0, stream>>>(X, psum, psq, Mrows, Cc);
    k_bn_finalize<<<cdiv(Cc, 256), 256, 0, stream>>>(psum, psq, gam, bet,
                                                     scale, shift, Cc,
                                                     1.0f / (float)Mrows);
  };

  // ---- stem: conv1 + bn0 + relu ----
  {
    const size_t total = (size_t)32 * 224 * 224 * 32;
    k_conv1<<<cdiv(total, 256), 256, 0, stream>>>(x, conv1_w, conv1_b, slot0,
                                                  32, 224, 224);
    stats_bn(slot0, 32 * 224 * 224, 32, bn0_g, bn0_b);
    k_bnrelu_f32<<<cdiv(total, 256), 256, 0, stream>>>(slot0, scale, shift,
                                                       total, 31);
  }

  // ---- 13 depthwise-separable blocks ----
  float* act = slot0;
  float* oth = slot1;
  int Hin = 224;
  for (int i = 0; i < 13; ++i) {
    const int Cin = CH[i], Cout = CH[i + 1], s = ST[i];
    const int Hout = (s == 2) ? (Hin / 2) : Hin;
    const size_t Mout = (size_t)32 * Hout * Hout;
    const size_t dtot = Mout * Cin;

    // depthwise 3x3 (shared kernel) + bias
    k_dwconv<<<cdiv(dtot, 256), 256, 0, stream>>>(act, oth, dw_w[i], dw_b[i],
                                                  32, Hin, Hin, Hout, Hout,
                                                  log2i(Cin), s);
    // bn1 + relu -> bf16 GEMM A (into the now-free "act" slot)
    stats_bn(oth, (int)Mout, Cin, bn1_g[i], bn1_b[i]);
    __bf16* abf = (__bf16*)act;
    k_bnrelu_bf16<<<cdiv(dtot, 256), 256, 0, stream>>>(oth, abf, scale, shift,
                                                       dtot, Cin - 1);
    // pointwise conv as bf16 WMMA GEMM
    const size_t ptot = (size_t)(Cin >> 5) * (Cout >> 4) * 512;
    k_pack_b<<<cdiv(ptot, 256), 256, 0, stream>>>(pw_w[i], packB, Cin, Cout, Cout);
    dim3 gg((unsigned)(Mout / 32), (unsigned)(Cout / 64));
    k_gemm_bf16<<<gg, 128, 0, stream>>>(abf, packB, pw_b[i], oth,
                                        (int)Mout, Cin, Cout, Cout);
    // bn2 + relu (in place, f32) -> next block input
    stats_bn(oth, (int)Mout, Cout, bn2_g[i], bn2_b[i]);
    const size_t gtot = Mout * (size_t)Cout;
    k_bnrelu_f32<<<cdiv(gtot, 256), 256, 0, stream>>>(oth, scale, shift,
                                                      gtot, Cout - 1);
    act = oth;
    oth = (act == slot0) ? slot1 : slot0;
    Hin = Hout;
  }

  // ---- avg pool + FC (WMMA GEMM, N padded 1000 -> 1024, stores guarded) ----
  k_avgpool<<<cdiv(32 * 1024, 256), 256, 0, stream>>>(act, pooled, 32, 49);
  const size_t ptot = (size_t)(1024 >> 5) * (1024 >> 4) * 512;
  k_pack_b<<<cdiv(ptot, 256), 256, 0, stream>>>(fc_w, packB, 1024, 1024, 1000);
  dim3 gfc(1u, 1024u / 64u);
  k_gemm_bf16<<<gfc, 128, 0, stream>>>(pooled, packB, fc_b, (float*)d_out,
                                       32, 1024, 1024, 1000);
}
